// WindowRouting_35107062677736
// MI455X (gfx1250) — matrix operand order repaired and checked by reading, compiled, not run on hardware
//
#include <hip/hip_runtime.h>

// ---------------------------------------------------------------------------
// WindowRouting top-4 router for MI455X (gfx1250, wave32).
//
//   scores = Q @ image^T   (scale dropped: monotonic, doesn't change top-k)
//   out    = per-query top-4 image indices (int32), descending by score.
//
// Compute-bound (AI ~1400 FLOP/B) -> v_wmma_f32_16x16x32_f16.
// Transposed mapping D[img_row][query] so each lane accumulates top-4 for one
// query in registers; final lane<->lane+16 merge via ds_swizzle SWAPX16.
//
// v3: sched_barrier pins all 16 ds_load_b128 A-fragment loads above the 8
// WMMAs so dscnt waits become incremental (no full drain per WMMA) and the
// two independent accumulation chains keep the matrix pipe busy.
// ---------------------------------------------------------------------------

typedef __attribute__((ext_vector_type(16))) _Float16 v16h;
typedef __attribute__((ext_vector_type(8)))  _Float16 v8h;
typedef __attribute__((ext_vector_type(4)))  _Float16 v4h;
typedef __attribute__((ext_vector_type(8)))  float    v8f;

#define DIMK    128
#define NQ      4096
#define NIMG    8192
#define NB      4
#define TOPW    4
#define CHUNK   64                 // image rows staged per LDS buffer
#define ROWH    (DIMK + 8)         // halves per LDS row (+8 halves kills bank conflicts)
#define WAVES   8
#define BLOCKQ  (16 * WAVES)       // 128 queries per workgroup
#define NCHUNK  (NIMG / CHUNK)     // 128

// Branchless sorted-insert of (v,idx) into descending top-4 registers.
#define TOP4_INSERT(v, idx) do {                                          \
    bool g0 = (v) > tv0, g1 = (v) > tv1, g2 = (v) > tv2, g3 = (v) > tv3;  \
    tv3 = g2 ? tv2 : (g3 ? (v)   : tv3);                                  \
    ti3 = g2 ? ti2 : (g3 ? (idx) : ti3);                                  \
    tv2 = g1 ? tv1 : (g2 ? (v)   : tv2);                                  \
    ti2 = g1 ? ti1 : (g2 ? (idx) : ti2);                                  \
    tv1 = g0 ? tv0 : (g1 ? (v)   : tv1);                                  \
    ti1 = g0 ? ti0 : (g1 ? (idx) : ti1);                                  \
    tv0 = g0 ? (v)   : tv0;                                               \
    ti0 = g0 ? (idx) : ti0;                                               \
} while (0)

__device__ __forceinline__ float swzf16(float x) {
    // ds_swizzle_b32 group-of-32, xor_mask=0x10 (SWAPX16): lane l <-> l^16
    return __int_as_float(__builtin_amdgcn_ds_swizzle(__float_as_int(x), 0x401f));
}
__device__ __forceinline__ int swzi16(int x) {
    return __builtin_amdgcn_ds_swizzle(x, 0x401f);
}

// A-tile fragment load from LDS. contiguous k order chosen here (B uses the same):
//   lane<16 : halves [32w+0..15] of row (l16)
//   lane>=16: halves [32w+16..31]
__device__ __forceinline__ v16h load_a(const _Float16* __restrict__ L,
                                       int row, int w, bool hi) {
    const _Float16* __restrict__ ar = L + row * ROWH + 32 * w + (hi ? 16 : 0);
    v8h a0 = *(const v8h*)ar;
    v8h a1 = *(const v8h*)(ar + 8);
    return __builtin_shufflevector(a0, a1,
        0, 1, 2, 3, 4, 5, 6, 7, 8, 9, 10, 11, 12, 13, 14, 15);
}

// Cooperative f32 -> f16 stage of CHUNK image rows into an LDS buffer.
// 256 threads: 4 threads per row, 32 columns each (8x float4 -> 8x v4h).
__device__ __forceinline__ void fill_chunk(_Float16* __restrict__ buf,
                                           const float* __restrict__ imgrows,
                                           int tid) {
    const int row = tid >> 2;
    const int col = (tid & 3) * 32;
    const float4* __restrict__ src =
        (const float4*)(imgrows + (size_t)row * DIMK + col);
    _Float16* __restrict__ dst = buf + row * ROWH + col;
#pragma unroll
    for (int i = 0; i < 8; ++i) {
        float4 f = src[i];
        v4h h;
        h[0] = (_Float16)f.x; h[1] = (_Float16)f.y;
        h[2] = (_Float16)f.z; h[3] = (_Float16)f.w;
        *(v4h*)(dst + 4 * i) = h;
    }
}

__global__ __launch_bounds__(256) void window_routing_top4(
    const float* __restrict__ query,   // (4, 4096, 128) f32
    const float* __restrict__ image,   // (4, 8192, 128) f32
    int* __restrict__ out)             // (4, 4096, 4) int32
{
    __shared__ __align__(16) _Float16 lds[2][CHUNK * ROWH];

    const int  tid  = threadIdx.x;
    const int  lane = tid & 31;
    const int  wave = tid >> 5;
    const int  l16  = lane & 15;
    const bool hi   = lane >= 16;

    const int batch = blockIdx.x >> 5;        // 32 blocks per batch
    const int qbase = (blockIdx.x & 31) * BLOCKQ;
    const int qrow  = qbase + wave * 16 + l16; // query row owned by this lane pair

    // ---- B matrix (Q^T, K=128 x N=16) registers, loop-invariant per wave ----
    // B uses the same contiguous k order as A: lanes 0-15 hold K=0..15, lanes 16-31 hold K=16..31,
    // 2 K-values packed per VGPR -> lane reads 16 contiguous halves of row qrow.
    const float* __restrict__ qp = query + ((size_t)batch * NQ + qrow) * DIMK;
    v16h bq[4];
#pragma unroll
    for (int w = 0; w < 4; ++w) {
        const float4* __restrict__ p4 = (const float4*)(qp + 32 * w + (hi ? 16 : 0));
        float4 f[4];
#pragma unroll
        for (int i = 0; i < 4; ++i) f[i] = p4[i];
        v16h t;
#pragma unroll
        for (int i = 0; i < 4; ++i) {
            t[4 * i + 0] = (_Float16)f[i].x;
            t[4 * i + 1] = (_Float16)f[i].y;
            t[4 * i + 2] = (_Float16)f[i].z;
            t[4 * i + 3] = (_Float16)f[i].w;
        }
        bq[w] = t;
    }

    float tv0 = -1e30f, tv1 = -1e30f, tv2 = -1e30f, tv3 = -1e30f;
    int   ti0 = 0,      ti1 = 0,      ti2 = 0,      ti3 = 0;

    const float* __restrict__ imgb = image + (size_t)batch * NIMG * DIMK;

    // ---- software-pipelined chunk loop (double-buffered LDS) ----
    fill_chunk(&lds[0][0], imgb, tid);
    __syncthreads();

    for (int c = 0; c < NCHUNK; ++c) {
        const int buf = c & 1;
        if (c + 1 < NCHUNK) {
            fill_chunk(&lds[buf ^ 1][0], imgb + (size_t)(c + 1) * CHUNK * DIMK, tid);
            if (c + 2 < NCHUNK) {
                // Warm GL2 for the chunk after next (global_prefetch_b8).
                const float* nxt = imgb + (size_t)(c + 2) * CHUNK * DIMK
                                 + (size_t)(tid >> 2) * DIMK + (tid & 3) * 32;
                __builtin_prefetch(nxt, 0, 1);
            }
        }

        const _Float16* __restrict__ L = &lds[buf][0];
#pragma unroll
        for (int sp = 0; sp < CHUNK / 32; ++sp) {  // two 16-row sub-tiles per pass
            // Preload all 8 A fragments into distinct registers ...
            v16h a0[4], a1[4];
#pragma unroll
            for (int w = 0; w < 4; ++w) {
                a0[w] = load_a(L, sp * 32 + l16,      w, hi);
                a1[w] = load_a(L, sp * 32 + 16 + l16, w, hi);
            }
            // ... and pin them above the WMMA burst so dscnt waits are
            // incremental instead of a full drain before every WMMA.
            __builtin_amdgcn_sched_barrier(0);

            // Two independent accumulation chains -> matrix pipe stays busy.
            v8f acc0 = {}, acc1 = {};
#pragma unroll
            for (int w = 0; w < 4; ++w) {
                acc0 = __builtin_amdgcn_wmma_f32_16x16x32_f16(
                    false, a0[w], false, bq[w], (short)0, acc0, false, false);
                acc1 = __builtin_amdgcn_wmma_f32_16x16x32_f16(
                    false, a1[w], false, bq[w], (short)0, acc1, false, false);
            }

            // D layout: lane owns query l16; VGPR j = image row base + j (+8 if hi)
            const int ib0 = c * CHUNK + sp * 32 + (hi ? 8 : 0);
            const int ib1 = ib0 + 16;
            bool beat = false;
#pragma unroll
            for (int j = 0; j < 8; ++j) beat |= (acc0[j] > tv3) | (acc1[j] > tv3);
            if (__builtin_amdgcn_ballot_w32(beat)) {   // uniform skip: inserts are rare
#pragma unroll
                for (int j = 0; j < 8; ++j) { TOP4_INSERT(acc0[j], ib0 + j); }
#pragma unroll
                for (int j = 0; j < 8; ++j) { TOP4_INSERT(acc1[j], ib1 + j); }
            }
        }
        __syncthreads();
    }

    // ---- merge lane l (rows m=0..7 of each subtile) with lane l+16 (m=8..15) ----
    float ov0 = swzf16(tv0), ov1 = swzf16(tv1), ov2 = swzf16(tv2), ov3 = swzf16(tv3);
    int   oi0 = swzi16(ti0), oi1 = swzi16(ti1), oi2 = swzi16(ti2), oi3 = swzi16(ti3);
    TOP4_INSERT(ov0, oi0);
    TOP4_INSERT(ov1, oi1);
    TOP4_INSERT(ov2, oi2);
    TOP4_INSERT(ov3, oi3);

    if (!hi) {
        int* __restrict__ o = out + ((size_t)batch * NQ + qrow) * TOPW;
        o[0] = ti0; o[1] = ti1; o[2] = ti2; o[3] = ti3;
    }
}

extern "C" void kernel_launch(void* const* d_in, const int* in_sizes, int n_in,
                              void* d_out, int out_size, void* d_ws, size_t ws_size,
                              hipStream_t stream) {
    (void)in_sizes; (void)n_in; (void)out_size; (void)d_ws; (void)ws_size;
    const float* query = (const float*)d_in[0];
    const float* image = (const float*)d_in[1];
    int* out = (int*)d_out;

    dim3 grid(NB * (NQ / BLOCKQ));   // 4 batches * 32 query-blocks = 128
    dim3 block(256);                 // 8 wave32 waves
    window_routing_top4<<<grid, block, 0, stream>>>(query, image, out);
}
